// GraphConvGrapHiC_21492016349660
// MI455X (gfx1250) — matrix operand order repaired and checked by reading, not compiled, not run on hardware
//
#include <hip/hip_runtime.h>
#include <hip/hip_bf16.h>
#include <math.h>

// ---------- types for WMMA ----------
typedef __attribute__((ext_vector_type(16))) __bf16 v16bf;
typedef __attribute__((ext_vector_type(8)))  float  v8f;

union ABu { v16bf v; unsigned short u[16]; };

__device__ __forceinline__ unsigned short f2bf(float f) {
    union { float f; unsigned u; } x; x.f = f;
    unsigned r = x.u + 0x7FFFu + ((x.u >> 16) & 1u);   // round-to-nearest-even
    return (unsigned short)(r >> 16);
}
__device__ __forceinline__ float bf2f(unsigned short b) {
    union { unsigned u; float f; } x; x.u = ((unsigned)b) << 16;
    return x.f;
}

#define NT 4096     // total nodes
#define NN 256      // nodes per graph
#define NB 16       // graphs
#define NE 131072   // edges
#define TILE 32     // output tile (i and j)
#define HALO 38     // TILE + 6
#define CBLK 48     // padded compute range (3 x 16)
#define PIX  (HALO * HALO)      // 1444 halo pixels
#define MT   ((PIX + 15) / 16)  // 91 M-tiles for the conv GEMM

// ---------- GCN helper kernels ----------
__global__ void k_zero(float* p, int n) {
    int i = blockIdx.x * blockDim.x + threadIdx.x;
    if (i < n) p[i] = 0.f;
}

__global__ void k_deg(const int* __restrict__ ei, const float* __restrict__ ew,
                      float* __restrict__ deg) {
    int e = blockIdx.x * blockDim.x + threadIdx.x;
    if (e < NE) atomicAdd(&deg[ei[NE + e]], ew[e]);
}

__global__ void k_dinv(float* p) {                     // deg -> rsqrt(deg + 2)
    int i = blockIdx.x * blockDim.x + threadIdx.x;
    if (i < NT) p[i] = rsqrtf(p[i] + 2.0f);            // self-loop weight 2.0, always > 0
}

// out[n,o] = bias[o] + 2*dinv[n]^2 * t[n,o]   (self-loop contribution + init)
__global__ void k_init_self(const float* __restrict__ t, const float* __restrict__ bias,
                            const float* __restrict__ dinv, float* __restrict__ out, int O) {
    int i = blockIdx.x * blockDim.x + threadIdx.x;
    if (i < NT * O) {
        int n = i / O, o = i - n * O;
        float di = dinv[n];
        out[i] = bias[o] + 2.0f * di * di * t[i];
    }
}

__global__ void k_edge(const int* __restrict__ ei, const float* __restrict__ ew,
                       const float* __restrict__ dinv, const float* __restrict__ t,
                       float* __restrict__ out, int O) {
    int e = blockIdx.x * blockDim.x + threadIdx.x;
    if (e >= NE) return;
    int s = ei[e], d = ei[NE + e];
    float norm = dinv[s] * ew[e] * dinv[d];
    const float* ts = t + (size_t)s * O;
    float* od = out + (size_t)d * O;
    for (int o = 0; o < O; ++o) atomicAdd(&od[o], norm * ts[o]);
}

__global__ void k_relu(float* p, int n) {
    int i = blockIdx.x * blockDim.x + threadIdx.x;
    if (i < n) p[i] = fmaxf(p[i], 0.f);
}

// ---------- dense GEMM via WMMA:  T[16-row tile, 16-col tile] = A(K) x W(KxO) ----------
__global__ __launch_bounds__(32) void k_gemm_wmma(const float* __restrict__ A,
                                                  const float* __restrict__ W,
                                                  float* __restrict__ T, int K, int O) {
    int lane = threadIdx.x & 31;
    int m    = lane & 15;
    int half = lane >> 4;
    int row0 = blockIdx.x * 16;
    int col0 = blockIdx.y * 16;

    ABu a, b;
    #pragma unroll
    for (int e = 0; e < 16; ++e) {
        int ka = (half << 4) + e;                        // A: K 0-15 | 16-31
        a.u[e] = (ka < K) ? f2bf(A[(row0 + m) * K + ka]) : (unsigned short)0;
        int kb = (half << 4) + e;                        // B: K 0-15 | 16-31
        b.u[e] = (kb < K) ? f2bf(W[kb * O + col0 + m])  : (unsigned short)0;
    }
    v8f c = {};
    v8f d = __builtin_amdgcn_wmma_f32_16x16x32_bf16(false, a.v, false, b.v,
                                                    (short)0, c, false, false);
    #pragma unroll
    for (int r = 0; r < 8; ++r)
        T[(row0 + r + half * 8) * O + col0 + m] = d[r];
}

// ---------- fused ContactCNN kernel ----------
// Phase 1: c halo (38x38 x 32ch) in LDS via WMMA (outer product + 1x1 conv + bias + relu)
// Phase 2: conv as GEMM  T[pixel, tap] = c[pixel, h] x Wp[h, tap]  via WMMA (4 tap-chunks)
// Phase 3: 49-term gather-add per output pixel + bias + sigmoid
__global__ __launch_bounds__(256) void k_contact(const float* __restrict__ z,
                                                 const float* __restrict__ Wc,
                                                 const float* __restrict__ bc,
                                                 const float* __restrict__ Wp,
                                                 const float* __restrict__ bp,
                                                 float* __restrict__ out) {
    __shared__ unsigned short c_s[PIX * 32];        // 92,416 B  bf16 c, [pixel][h]
    __shared__ unsigned short T_s[PIX * 16];        // 46,208 B  bf16 T chunk, [pixel][n]
    __shared__ float zi_s[CBLK * 32];               //  6,144 B  z rows, i-halo (zero pad)
    __shared__ float zj_s[CBLK * 32];               //  6,144 B  z rows, j-halo
    __shared__ float wc_s[32 * 32];                 //  4,096 B  Wc[h][d]
    __shared__ unsigned short wpT_s[32 * 64];       //  4,096 B  bf16 Wp [h][tap], pad->64
    __shared__ float bc_s[32];

    const int tid = threadIdx.x;
    const int b   = blockIdx.z;
    const int i0  = blockIdx.y * TILE;
    const int j0  = blockIdx.x * TILE;

    // ---- stage: z halos via gfx1250 async global->LDS copies (ASYNCcnt) ----
    for (int idx = tid; idx < CBLK * 32; idx += 256) {
        int ii = idx >> 5, d = idx & 31;
        int gi = i0 - 3 + ii;
        if (gi >= 0 && gi < NN) {
            unsigned long long ga =
                (unsigned long long)(uintptr_t)&z[((size_t)b * NN + gi) * 32 + d];
            unsigned la = (unsigned)(uintptr_t)&zi_s[idx];
            asm volatile("global_load_async_to_lds_b32 %0, %1, off"
                         :: "v"(la), "v"(ga) : "memory");
        } else {
            zi_s[idx] = 0.f;
        }
        int gj = j0 - 3 + ii;
        if (gj >= 0 && gj < NN) {
            unsigned long long ga =
                (unsigned long long)(uintptr_t)&z[((size_t)b * NN + gj) * 32 + d];
            unsigned la = (unsigned)(uintptr_t)&zj_s[idx];
            asm volatile("global_load_async_to_lds_b32 %0, %1, off"
                         :: "v"(la), "v"(ga) : "memory");
        } else {
            zj_s[idx] = 0.f;
        }
    }
    for (int idx = tid; idx < 32 * 32; idx += 256) wc_s[idx] = Wc[idx];
    for (int idx = tid; idx < 32 * 64; idx += 256) {
        int h = idx >> 6, q = idx & 63;
        wpT_s[idx] = (q < 49) ? f2bf(Wp[h * 49 + q]) : (unsigned short)0;
    }
    if (tid < 32) bc_s[tid] = bc[tid];
    asm volatile("s_wait_asynccnt 0" ::: "memory");
    __syncthreads();

    const int wave = tid >> 5;
    const int lane = tid & 31;
    const int m    = lane & 15;
    const int half = lane >> 4;

    // ---- phase 1: c = relu( (z_i . Wc[h]) x z_j^T + bc[h] ), 288 WMMA over 8 waves ----
    for (int t = 0; t < 36; ++t) {
        int item = wave * 36 + t;          // 0..287
        int h    = item & 31;
        int blk  = item >> 5;              // 0..8
        int bi   = blk / 3, bj = blk - 3 * (blk / 3);

        ABu a, bb;
        #pragma unroll
        for (int e = 0; e < 16; ++e) {
            int ka = (half << 4) + e;
            a.u[e]  = f2bf(zi_s[(bi * 16 + m) * 32 + ka] * wc_s[h * 32 + ka]);
            int kb = (half << 4) + e;
            bb.u[e] = f2bf(zj_s[(bj * 16 + m) * 32 + kb]);
        }
        v8f c;
        #pragma unroll
        for (int r = 0; r < 8; ++r) c[r] = bc_s[h];
        v8f d = __builtin_amdgcn_wmma_f32_16x16x32_bf16(false, a.v, false, bb.v,
                                                        (short)0, c, false, false);
        #pragma unroll
        for (int r = 0; r < 8; ++r) {
            int ii = bi * 16 + r + half * 8;
            int jj = bj * 16 + m;
            if (ii < HALO && jj < HALO) {
                int gi = i0 - 3 + ii, gj = j0 - 3 + jj;
                // SAME padding: conv input is 0 outside the tensor (NOT relu(bc))
                float v = (gi >= 0 && gi < NN && gj >= 0 && gj < NN)
                              ? fmaxf(d[r], 0.f) : 0.f;
                c_s[(ii * HALO + jj) * 32 + h] = f2bf(v);
            }
        }
    }
    __syncthreads();

    // ---- phase 2+3: conv-as-GEMM over 4 tap-chunks, then gather-add ----
    float acc[4];
    {
        float bpv = bp[0];
        acc[0] = bpv; acc[1] = bpv; acc[2] = bpv; acc[3] = bpv;
    }
    for (int pass = 0; pass < 4; ++pass) {
        // T chunk: [pixel 0..1443] x [tap pass*16 .. pass*16+15]
        for (int tile = wave; tile < MT; tile += 8) {
            int rowbase = tile * 16;
            int pr = rowbase + m; if (pr > PIX - 1) pr = PIX - 1;   // clamp pad rows
            ABu a, bb;
            #pragma unroll
            for (int e = 0; e < 16; ++e) {
                int ka = (half << 4) + e;
                a.u[e]  = c_s[pr * 32 + ka];                        // already bf16
                int kb = (half << 4) + e;
                bb.u[e] = wpT_s[kb * 64 + pass * 16 + m];
            }
            v8f c = {};
            v8f d = __builtin_amdgcn_wmma_f32_16x16x32_bf16(false, a.v, false, bb.v,
                                                            (short)0, c, false, false);
            #pragma unroll
            for (int r = 0; r < 8; ++r) {
                int row = rowbase + r + half * 8;
                int q   = pass * 16 + m;
                if (row < PIX && q < 49)
                    T_s[row * 16 + m] = f2bf(d[r]);
            }
        }
        __syncthreads();

        // gather-add: out(oi,oj) += sum_n T[(oi+di)*38 + oj+dj, n]
        int nv = (pass == 3) ? 1 : 16;        // taps 48..63 are padding except q=48
        #pragma unroll
        for (int pp = 0; pp < 4; ++pp) {
            int p  = tid + pp * 256;
            int oi = p >> 5, oj = p & 31;
            float s = 0.f;
            for (int n = 0; n < nv; ++n) {
                int q  = pass * 16 + n;
                int di = q / 7, dj = q - 7 * di;
                s += bf2f(T_s[((oi + di) * HALO + (oj + dj)) * 16 + n]);
            }
            acc[pp] += s;
        }
        __syncthreads();                       // T_s reused next pass
    }

    // ---- sigmoid + store ----
    #pragma unroll
    for (int pp = 0; pp < 4; ++pp) {
        int p  = tid + pp * 256;
        int oi = p >> 5, oj = p & 31;
        float s = 1.f / (1.f + __expf(-acc[pp]));
        out[((size_t)b * NN + (i0 + oi)) * NN + (j0 + oj)] = s;
    }
}

// ---------- host launcher ----------
extern "C" void kernel_launch(void* const* d_in, const int* in_sizes, int n_in,
                              void* d_out, int out_size, void* d_ws, size_t ws_size,
                              hipStream_t stream) {
    const float* x   = (const float*)d_in[0];
    const int*   ei  = (const int*)  d_in[1];   // [2, E]
    const float* ea  = (const float*)d_in[2];
    // d_in[3] = batch_index (contiguous reshape, unused)
    const float* W0  = (const float*)d_in[4];
    const float* b0  = (const float*)d_in[5];
    const float* W1  = (const float*)d_in[6];
    const float* b1  = (const float*)d_in[7];
    const float* W2  = (const float*)d_in[8];
    const float* b2  = (const float*)d_in[9];
    const float* Wc  = (const float*)d_in[10];
    const float* bc  = (const float*)d_in[11];
    const float* Wp  = (const float*)d_in[12];
    const float* bp  = (const float*)d_in[13];
    float* out = (float*)d_out;

    float* dinv = (float*)d_ws;          // NT
    float* t    = dinv + NT;             // NT*32
    float* ha   = t    + NT * 32;        // NT*32
    float* hb   = ha   + NT * 32;        // NT*32

    // degree -> dinv
    k_zero<<<NT / 256, 256, 0, stream>>>(dinv, NT);
    k_deg <<<NE / 256, 256, 0, stream>>>(ei, ea, dinv);
    k_dinv<<<NT / 256, 256, 0, stream>>>(dinv);

    // layer 0: x(32) -> ha(16)
    k_gemm_wmma<<<dim3(NT / 16, 1), 32, 0, stream>>>(x, W0, t, 32, 16);
    k_init_self<<<NT * 16 / 256, 256, 0, stream>>>(t, b0, dinv, ha, 16);
    k_edge<<<NE / 256, 256, 0, stream>>>(ei, ea, dinv, t, ha, 16);
    k_relu<<<NT * 16 / 256, 256, 0, stream>>>(ha, NT * 16);

    // layer 1: ha(16) -> hb(32)
    k_gemm_wmma<<<dim3(NT / 16, 2), 32, 0, stream>>>(ha, W1, t, 16, 32);
    k_init_self<<<NT * 32 / 256, 256, 0, stream>>>(t, b1, dinv, hb, 32);
    k_edge<<<NE / 256, 256, 0, stream>>>(ei, ea, dinv, t, hb, 32);
    k_relu<<<NT * 32 / 256, 256, 0, stream>>>(hb, NT * 32);

    // layer 2: hb(32) -> ha(32)  (ha becomes z)
    k_gemm_wmma<<<dim3(NT / 16, 2), 32, 0, stream>>>(hb, W2, t, 32, 32);
    k_init_self<<<NT * 32 / 256, 256, 0, stream>>>(t, b2, dinv, ha, 32);
    k_edge<<<NE / 256, 256, 0, stream>>>(ei, ea, dinv, t, ha, 32);
    k_relu<<<NT * 32 / 256, 256, 0, stream>>>(ha, NT * 32);

    // fused outer-product + 1x1 conv + relu + conv-as-GEMM 7x7 + sigmoid
    k_contact<<<dim3(NN / TILE, NN / TILE, NB), 256, 0, stream>>>(ha, Wc, bc, Wp, bp, out);
}
